// DreamerRSSM_52020643889844
// MI455X (gfx1250) — compile-verified, matrix operand order checked by reading, not run on hardware
//
#include <hip/hip_runtime.h>
#include <hip/hip_bf16.h>
#include <stdint.h>
#include <stddef.h>

// ---------------------------------------------------------------------------
// Dreamer RSSM forward scan for MI455X (gfx1250).
// One workgroup = 16 batch rows, loops over all T=64 steps.
// - GEMMs: v_wmma_f32_16x16x32_bf16, weights pre-packed into fragment-native
//   bf16 tiles (wave reads one contiguous 1KB block per B-fragment from L2).
// - embeds[:,t,:] staged LDS-ward with global_load_async_to_lds_b128 (ASYNCcnt)
//   issued at the top of each step, drained just before the posterior head.
// - Biases folded into LN / unimix consumers (no WMMA->VALU add hazard in
//   GEMM epilogues).
// ---------------------------------------------------------------------------

typedef __bf16 bf16;
typedef __attribute__((ext_vector_type(16))) __bf16 v16bf;
typedef __attribute__((ext_vector_type(8)))  __bf16 v8bf;
typedef __attribute__((ext_vector_type(8)))  float  v8f;

#define BB   512
#define TT   64
#define AA   32
#define DD   512
#define HH   512
#define SD   1024
#define GRUK 1536   // SD + H  (GRU input width, also xh row stride)
#define STATEW 1536 // D + SD

#define WMMA_BF16(a, b, c) \
  __builtin_amdgcn_wmma_f32_16x16x32_bf16(false, (a), false, (b), (short)0, (c), false, false)

#define V8F_ZERO {0.f, 0.f, 0.f, 0.f, 0.f, 0.f, 0.f, 0.f}

__device__ __forceinline__ uint32_t hash3(uint32_t a, uint32_t b, uint32_t c) {
  uint32_t h = a * 0x9E3779B1u ^ b * 0x85EBCA77u ^ c * 0xC2B2AE3Du ^ 0x27220A95u;
  h ^= h >> 16; h *= 0x7FEB352Du; h ^= h >> 15; h *= 0x846CA68Bu; h ^= h >> 16;
  return h;
}

__device__ __forceinline__ float sigm(float x) { return 1.f / (1.f + __expf(-x)); }
__device__ __forceinline__ float mishf(float x) {
  float sp = (x > 20.f) ? x : log1pf(__expf(x));
  return x * tanhf(sp);
}

// Async DMA: one 16-byte chunk global -> LDS, tracked by ASYNCcnt.
__device__ __forceinline__ void async_b128_to_lds(uint32_t lds_off,
                                                  const void* gaddr) {
  asm volatile("global_load_async_to_lds_b128 %0, %1, off"
               :: "v"(lds_off), "v"((uint64_t)(uintptr_t)gaddr)
               : "memory");
}
__device__ __forceinline__ void wait_asynccnt0() {
  asm volatile("s_wait_asynccnt 0" ::: "memory");
}

// A-fragment (activations) from row-major LDS per the CDNA5 16-bit layout:
// lanes 0-15 hold K = {0..7, 16..23}, lanes 16-31 hold K = {8..15, 24..31}.
__device__ __forceinline__ v16bf load_frag(const bf16* __restrict__ base,
                                           int row, int ld, int k0, int lane) {
  const bf16* p = base + (size_t)row * ld + k0 + ((lane >> 4) << 3);
  v8bf lo = *(const v8bf*)(p);
  v8bf hi = *(const v8bf*)(p + 16);
  return __builtin_shufflevector(lo, hi, 0, 1, 2, 3, 4, 5, 6, 7,
                                 8, 9, 10, 11, 12, 13, 14, 15);
}

// B-fragment from packed weights: tile (jt, ks) is a contiguous 512-element
// (1KB) block, lane-major, each lane's 16 bf16 contiguous.
__device__ __forceinline__ v16bf load_packed(const bf16* __restrict__ Wp,
                                             int tile, int ks, int ksteps,
                                             int lane) {
  const bf16* p = Wp + ((((size_t)tile * ksteps) + ks) * 32 + lane) * 16;
  v8bf lo = *(const v8bf*)(p);
  v8bf hi = *(const v8bf*)(p + 8);
  return __builtin_shufflevector(lo, hi, 0, 1, 2, 3, 4, 5, 6, 7,
                                 8, 9, 10, 11, 12, 13, 14, 15);
}

// GEMM: scr[16][512] = A[16][K](bf16,LDS) @ Wp(tiles tile_ofs..)^T  (no bias;
// bias folded into consumers). 8 waves x 4 N-tiles, k-outer, 4 persistent
// accumulators so each LDS A-fragment feeds 4 WMMAs.
__device__ __forceinline__ void gemm_to_scr(
    float scr[16][512], const bf16* __restrict__ A, int lda,
    const bf16* __restrict__ Wp, int K, int tile_ofs, int lane, int wave) {
  const int arow = lane & 15;
  const int ksteps = K >> 5;
  v8f acc0 = V8F_ZERO, acc1 = V8F_ZERO, acc2 = V8F_ZERO, acc3 = V8F_ZERO;
  for (int ks = 0; ks < ksteps; ++ks) {
    v16bf a = load_frag(A, arow, lda, ks * 32, lane);
    acc0 = WMMA_BF16(a, load_packed(Wp, tile_ofs + wave,      ks, ksteps, lane), acc0);
    acc1 = WMMA_BF16(a, load_packed(Wp, tile_ofs + wave + 8,  ks, ksteps, lane), acc1);
    acc2 = WMMA_BF16(a, load_packed(Wp, tile_ofs + wave + 16, ks, ksteps, lane), acc2);
    acc3 = WMMA_BF16(a, load_packed(Wp, tile_ofs + wave + 24, ks, ksteps, lane), acc3);
  }
  v8f accs[4] = {acc0, acc1, acc2, acc3};
  #pragma unroll
  for (int i = 0; i < 4; ++i) {
    int ncol = (wave + i * 8) * 16 + (lane & 15);
    #pragma unroll
    for (int v = 0; v < 8; ++v) {
      int m = v + ((lane >> 4) << 3);
      scr[m][ncol] = accs[i][v];
    }
  }
}

// Row-wise LayerNorm (population var, eps 1e-5) + Mish over (scr + bias),
// f32 scratch -> bf16 dst.
__device__ __forceinline__ void ln_mish_scr_to_bf(
    float scr[16][512], const float* __restrict__ bias,
    const float* __restrict__ g, const float* __restrict__ bt,
    bf16* __restrict__ dst, int ldd, float* mv, int tid) {
  if (tid < 16) {
    float s = 0.f, s2 = 0.f;
    for (int c = 0; c < 512; ++c) {
      float x = scr[tid][c] + bias[c];
      s += x; s2 += x * x;
    }
    float m = s * (1.f / 512.f);
    mv[tid]      = m;
    mv[16 + tid] = rsqrtf(fmaxf(s2 * (1.f / 512.f) - m * m, 0.f) + 1e-5f);
  }
  __syncthreads();
  for (int i = tid; i < 16 * 512; i += 256) {
    int r = i >> 9, c = i & 511;
    float x = (scr[r][c] + bias[c] - mv[r]) * mv[16 + r] * g[c] + bt[c];
    dst[(size_t)r * ldd + c] = (bf16)mishf(x);
  }
  __syncthreads();
}

// Unimix over one half (16 stoch groups of 32) of (scr + bias): softmax,
// mix with uniform, log, write logits to global; optionally leave probs in
// scratch for sampling.
__device__ __forceinline__ void unimix_half(
    float scr[16][512], const float* __restrict__ bias, int bias_ofs,
    float* __restrict__ outg, int b0, int t, int s_base, bool keep_probs,
    int tid) {
  for (int g = tid; g < 256; g += 256) {
    int m = g >> 4, sl = g & 15;
    float* p = &scr[m][sl * 32];
    const float* bb = bias + bias_ofs + sl * 32;
    float mx = -1e30f;
    for (int c = 0; c < 32; ++c) mx = fmaxf(mx, p[c] + bb[c]);
    float sum = 0.f;
    for (int c = 0; c < 32; ++c) sum += __expf(p[c] + bb[c] - mx);
    float inv = 1.f / sum;
    float* og = outg + ((size_t)(b0 + m) * TT + t) * SD + (size_t)(s_base + sl) * 32;
    for (int c = 0; c < 32; ++c) {
      float pr = 0.99f * __expf(p[c] + bb[c] - mx) * inv + (0.01f / 32.f);
      og[c] = __logf(fmaxf(pr, 1e-8f));
      if (keep_probs) p[c] = pr;
    }
  }
}

// Categorical sample (deterministic hash RNG) + straight-through one-hot.
__device__ __forceinline__ void sample_half(
    float scr[16][512], bf16 xh[16][GRUK], float* __restrict__ states,
    int b0, int t, int s_base, int tid) {
  for (int g = tid; g < 256; g += 256) {
    int m = g >> 4, sl = g & 15;
    int s = s_base + sl;
    float* p = &scr[m][sl * 32];
    float u = (hash3((uint32_t)(b0 + m), (uint32_t)t, (uint32_t)s) >> 8)
              * (1.f / 16777216.f);
    float ca = 0.f; int idx = 31;
    for (int c = 0; c < 32; ++c) { ca += p[c]; if (u < ca) { idx = c; break; } }
    float* so = states + ((size_t)(b0 + m) * TT + t) * STATEW + DD + (size_t)s * 32;
    for (int c = 0; c < 32; ++c) {
      float oh = (c == idx) ? 1.f : 0.f;
      xh[m][s * 32 + c] = (bf16)oh;   // next-step stoch (bf16 GEMM operand)
      so[c] = oh;                     // states[..., 512 + s*32 + c]
    }
  }
}

__global__ void __launch_bounds__(256, 1)
rssm_scan_kernel(
    const float* __restrict__ actions, const float* __restrict__ embeds,
    const float* __restrict__ ab,  const float* __restrict__ ag,
    const float* __restrict__ abt,
    const float* __restrict__ b_ih, const float* __restrict__ b_hh,
    const float* __restrict__ pr_b1, const float* __restrict__ pr_g,
    const float* __restrict__ pr_bt, const float* __restrict__ pr_b2,
    const float* __restrict__ po_b1, const float* __restrict__ po_g,
    const float* __restrict__ po_bt, const float* __restrict__ po_b2,
    const bf16* __restrict__ aw_p,  const bf16* __restrict__ wih_p,
    const bf16* __restrict__ whh_p, const bf16* __restrict__ prw1_p,
    const bf16* __restrict__ prw2_p, const bf16* __restrict__ pow1_p,
    const bf16* __restrict__ pow2_p,
    float* __restrict__ states, float* __restrict__ priors,
    float* __restrict__ posts) {

  __shared__ bf16  xh[16][GRUK];     // [stoch(1024) | a / hid(512)] bf16 A-operand
  __shared__ bf16  hbf[16][DD];      // deter (bf16)
  __shared__ bf16  actb[16][AA];     // prev_action (bf16)
  __shared__ float scr[16][512];     // f32 pre-activation / logits-half scratch
  __shared__ float estg[16][512];    // async-staged embeds[:,t,:]
  __shared__ float mv[32];           // LN mean / inv-std

  const int tid  = threadIdx.x;
  const int lane = tid & 31;
  const int wave = tid >> 5;
  const int b0   = blockIdx.x * 16;

  // zero-init carried state (deter=0, stoch=0)
  for (int i = tid; i < 16 * SD; i += 256) xh[i >> 10][i & 1023] = (bf16)0.f;
  for (int i = tid; i < 16 * DD; i += 256) hbf[i >> 9][i & 511]  = (bf16)0.f;
  __syncthreads();

  for (int t = 0; t < TT; ++t) {
    // ---- 0) kick off async DMA of embeds[:,t,:] into estg (drained at 4) ----
    for (int i = tid; i < 2048; i += 256) {            // 2048 x 16B chunks
      int r = i >> 7, c4 = (i & 127) * 4;
      uint32_t lds_off = (uint32_t)(size_t)&estg[r][c4];
      const float* gsrc = embeds + ((size_t)(b0 + r) * TT + t) * 512 + c4;
      async_b128_to_lds(lds_off, (const void*)gsrc);
    }

    // ---- 1) action projection: a = mish(ln(prev_action @ aw^T + ab)) ----
    for (int i = tid; i < 16 * AA; i += 256) {
      int r = i >> 5, c = i & 31;
      float av = (t == 0) ? 0.f
                          : actions[((size_t)(b0 + r) * TT + (t - 1)) * AA + c];
      actb[r][c] = (bf16)av;
    }
    __syncthreads();
    gemm_to_scr(scr, &actb[0][0], AA, aw_p, AA, 0, lane, wave);
    __syncthreads();
    ln_mish_scr_to_bf(scr, ab, ag, abt, &xh[0][SD], GRUK, mv, tid);

    // ---- 2) GRU: deter = gru([stoch, a], deter) ----
    float dreg[4][8];
    {
      const int arow = lane & 15;
      for (int it = 0; it < 4; ++it) {
        const int jt   = wave + it * 8;           // 0..31 N-tile of 512
        const int ncol = jt * 16 + (lane & 15);
        v8f ar = V8F_ZERO, az = V8F_ZERO, an = V8F_ZERO;
        v8f hr = V8F_ZERO, hz = V8F_ZERO, hn = V8F_ZERO;
        for (int ks = 0; ks < GRUK / 32; ++ks) {
          __builtin_prefetch(
              (const void*)(wih_p + (((size_t)jt * 48) + ks) * 512 + 4096), 0, 1);
          v16bf a = load_frag(&xh[0][0], arow, GRUK, ks * 32, lane);
          ar = WMMA_BF16(a, load_packed(wih_p, jt,      ks, 48, lane), ar);
          az = WMMA_BF16(a, load_packed(wih_p, 32 + jt, ks, 48, lane), az);
          an = WMMA_BF16(a, load_packed(wih_p, 64 + jt, ks, 48, lane), an);
        }
        for (int ks = 0; ks < DD / 32; ++ks) {
          v16bf a = load_frag(&hbf[0][0], arow, DD, ks * 32, lane);
          hr = WMMA_BF16(a, load_packed(whh_p, jt,      ks, 16, lane), hr);
          hz = WMMA_BF16(a, load_packed(whh_p, 32 + jt, ks, 16, lane), hz);
          hn = WMMA_BF16(a, load_packed(whh_p, 64 + jt, ks, 16, lane), hn);
        }
        const float bir = b_ih[ncol], biz = b_ih[DD + ncol], bin = b_ih[2 * DD + ncol];
        const float bhr = b_hh[ncol], bhz = b_hh[DD + ncol], bhn = b_hh[2 * DD + ncol];
        #pragma unroll
        for (int v = 0; v < 8; ++v) {
          int m = v + ((lane >> 4) << 3);
          float r  = sigm(ar[v] + bir + hr[v] + bhr);
          float z  = sigm(az[v] + biz + hz[v] + bhz);
          float nn = tanhf(an[v] + bin + r * (hn[v] + bhn));
          float h0 = (float)hbf[m][ncol];
          dreg[it][v] = (1.f - z) * nn + z * h0;
        }
      }
    }
    __syncthreads();  // all waves done reading old deter
    for (int it = 0; it < 4; ++it) {
      int ncol = (wave + it * 8) * 16 + (lane & 15);
      #pragma unroll
      for (int v = 0; v < 8; ++v) {
        int m = v + ((lane >> 4) << 3);
        float d = dreg[it][v];
        hbf[m][ncol] = (bf16)d;
        states[((size_t)(b0 + m) * TT + t) * STATEW + ncol] = d;
      }
    }
    __syncthreads();

    // ---- 3) prior head ----
    gemm_to_scr(scr, &hbf[0][0], DD, prw1_p, DD, 0, lane, wave);
    __syncthreads();
    ln_mish_scr_to_bf(scr, pr_b1, pr_g, pr_bt, &xh[0][SD], GRUK, mv, tid);
    for (int h = 0; h < 2; ++h) {
      gemm_to_scr(scr, &xh[0][SD], GRUK, prw2_p, HH, h * 32, lane, wave);
      __syncthreads();
      unimix_half(scr, pr_b2, h * 512, priors, b0, t, h * 16, false, tid);
      __syncthreads();
    }

    // ---- 4) posterior head: concat(deter, async-staged embed) ----
    wait_asynccnt0();     // each wave drains its own DMA issues
    __syncthreads();      // ... then everyone sees all of estg
    for (int i = tid; i < 16 * DD; i += 256) {
      int r = i >> 9, c = i & 511;
      xh[r][c]      = hbf[r][c];
      xh[r][DD + c] = (bf16)estg[r][c];
    }
    __syncthreads();
    gemm_to_scr(scr, &xh[0][0], GRUK, pow1_p, 1024, 0, lane, wave);
    __syncthreads();
    ln_mish_scr_to_bf(scr, po_b1, po_g, po_bt, &xh[0][SD], GRUK, mv, tid);
    for (int h = 0; h < 2; ++h) {
      gemm_to_scr(scr, &xh[0][SD], GRUK, pow2_p, HH, h * 32, lane, wave);
      __syncthreads();
      unimix_half(scr, po_b2, h * 512, posts, b0, t, h * 16, true, tid);
      __syncthreads();
      sample_half(scr, xh, states, b0, t, h * 16, tid);
      __syncthreads();
    }
  }
}

// ---------------------------------------------------------------------------
// Pack f32 row-major W[N][K] into fragment-native bf16 tiles:
// element index i = ((tile*ksteps + ks)*32 + lane)*16 + e, where tile covers
// rows n = tile*16 + (lane&15); lane>=16 takes the K-half {8..15, 24..31}.
__global__ void pack_weight(const float* __restrict__ src, bf16* __restrict__ dst,
                            int N, int K) {
  int i = blockIdx.x * blockDim.x + threadIdx.x;
  if (i >= N * K) return;
  int e      = i & 15;
  int lane   = (i >> 4) & 31;
  int rest   = i >> 9;              // tile*ksteps + ks
  int ksteps = K >> 5;
  int tile   = rest / ksteps;
  int ks     = rest - tile * ksteps;
  int n      = tile * 16 + (lane & 15);
  int khalf  = (lane >> 4) * 8;
  int k      = ks * 32 + ((e < 8) ? (khalf + e) : (16 + khalf + (e - 8)));
  dst[i] = (bf16)src[(size_t)n * K + k];
}

extern "C" void kernel_launch(void* const* d_in, const int* in_sizes, int n_in,
                              void* d_out, int out_size, void* d_ws, size_t ws_size,
                              hipStream_t stream) {
  const float* actions = (const float*)d_in[0];
  const float* embeds  = (const float*)d_in[1];
  const float* aw    = (const float*)d_in[2];
  const float* ab    = (const float*)d_in[3];
  const float* ag    = (const float*)d_in[4];
  const float* abt   = (const float*)d_in[5];
  const float* w_ih  = (const float*)d_in[6];
  const float* w_hh  = (const float*)d_in[7];
  const float* b_ih  = (const float*)d_in[8];
  const float* b_hh  = (const float*)d_in[9];
  const float* pr_w1 = (const float*)d_in[10];
  const float* pr_b1 = (const float*)d_in[11];
  const float* pr_g  = (const float*)d_in[12];
  const float* pr_bt = (const float*)d_in[13];
  const float* pr_w2 = (const float*)d_in[14];
  const float* pr_b2 = (const float*)d_in[15];
  const float* po_w1 = (const float*)d_in[16];
  const float* po_b1 = (const float*)d_in[17];
  const float* po_g  = (const float*)d_in[18];
  const float* po_bt = (const float*)d_in[19];
  const float* po_w2 = (const float*)d_in[20];
  const float* po_b2 = (const float*)d_in[21];

  // packed bf16 weight arena in workspace (~9.54 MB)
  bf16* ws = (bf16*)d_ws;
  size_t o = 0;
  bf16* aw_p   = ws + o; o += (size_t)512 * 32;
  bf16* wih_p  = ws + o; o += (size_t)1536 * 1536;
  bf16* whh_p  = ws + o; o += (size_t)1536 * 512;
  bf16* prw1_p = ws + o; o += (size_t)512 * 512;
  bf16* prw2_p = ws + o; o += (size_t)1024 * 512;
  bf16* pow1_p = ws + o; o += (size_t)512 * 1024;
  bf16* pow2_p = ws + o; o += (size_t)1024 * 512;
  if (ws_size < o * sizeof(bf16)) return;

  auto pk = [&](const float* s, bf16* d, int N, int K) {
    int n = N * K;
    pack_weight<<<(n + 255) / 256, 256, 0, stream>>>(s, d, N, K);
  };
  pk(aw,    aw_p,   512, 32);
  pk(w_ih,  wih_p,  1536, 1536);
  pk(w_hh,  whh_p,  1536, 512);
  pk(pr_w1, prw1_p, 512, 512);
  pk(pr_w2, prw2_p, 1024, 512);
  pk(po_w1, pow1_p, 512, 1024);
  pk(po_w2, pow2_p, 1024, 512);

  float* out    = (float*)d_out;
  float* states = out;                                   // [B,T,1536]
  float* priors = out + (size_t)BB * TT * STATEW;        // [B,T,32,32]
  float* posts  = priors + (size_t)BB * TT * SD;         // [B,T,32,32]

  rssm_scan_kernel<<<BB / 16, 256, 0, stream>>>(
      actions, embeds, ab, ag, abt, b_ih, b_hh,
      pr_b1, pr_g, pr_bt, pr_b2, po_b1, po_g, po_bt, po_b2,
      aw_p, wih_p, whh_p, prw1_p, prw2_p, pow1_p, pow2_p,
      states, priors, posts);
}
